// Attention_10514079940879
// MI455X (gfx1250) — compile-verified, matrix operand order checked by reading, not run on hardware
//
#include <hip/hip_runtime.h>

typedef __bf16 bf16;
typedef bf16  v16bf __attribute__((ext_vector_type(16)));
typedef float v8f   __attribute__((ext_vector_type(8)));
typedef unsigned int u32;
typedef unsigned int uint32x4 __attribute__((ext_vector_type(4)));
typedef int int32x4 __attribute__((ext_vector_type(4)));
typedef int int32x8 __attribute__((ext_vector_type(8)));

constexpr int Bz = 2, Ts = 2048, Dm = 1024, Nh = 16, Kh = 8, Hd = 128;
constexpr int Mrows = Bz * Ts;  // 4096 (b,t) rows

// ---------------------------------------------------------------- WMMA helper
__device__ __forceinline__ v8f wmma_bf16(v16bf a, v16bf b, v8f c) {
  // 8 args: (neg_a, A, neg_b, B, c_mod, C, reuse_a, reuse_b)
  return __builtin_amdgcn_wmma_f32_16x16x32_bf16(false, a, false, b, (short)0, c,
                                                 false, false);
}

// 16-bit operand fragment (CDNA5 A-matrix striping): lane L (row=L%16, hf=L/16)
// holds k = [hf*8, hf*8+8) and [16+hf*8, 16+hf*8+8) -> exactly two b128 loads.
__device__ __forceinline__ v16bf frag_b128(const bf16* __restrict__ p, int hf) {
  const uint4* q = (const uint4*)(p + hf * 8);
  union { uint4 u[2]; v16bf v; } t;
  t.u[0] = q[0];  // k = hf*8 .. hf*8+7
  t.u[1] = q[2];  // k = 16+hf*8 .. 16+hf*8+7   (16 bf16 = 2 uint4 ahead)
  return t.v;
}

// --------------------------------------------- CDNA5 async global->LDS copy
// GLOBAL_LOAD_ASYNC_TO_LDS_B128 (ASYNCcnt-tracked), GV addressing mode.
__device__ __forceinline__ void async_ld_b128(void* lds_ptr, const void* gptr) {
  u32 ldso = (u32)(uintptr_t)lds_ptr;  // aperture map: LDS offset is the pointer's low 32 bits
  asm volatile("global_load_async_to_lds_b128 %0, %1, off"
               :: "v"(ldso), "v"(gptr) : "memory");
}
__device__ __forceinline__ void wait_async0() {
  asm volatile("s_wait_asynccnt 0" ::: "memory");
}

// ----------------------------------------------- CDNA5 Tensor Data Mover (TDM)
// One descriptor-driven DMA of a 2D tile (tile_h rows x tile_w_dw dwords,
// global row stride row_stride_dw dwords) into LDS, padding +4 dwords (16B)
// after every 64 dwords (256B) -> LDS row stride 272B = 136 bf16.
// D# bitfields per CDNA5 ISA 8.3/8.4. Issued per-wave; tracked by TENSORcnt.
__device__ __forceinline__ void tdm_load_2d(void* lds_ptr, const void* gptr,
                                            u32 tile_w_dw, u32 tile_h,
                                            u32 row_stride_dw) {
  u32 lds = (u32)(uintptr_t)lds_ptr;
  unsigned long long ga = (unsigned long long)(uintptr_t)gptr;
  uint32x4 g0;
  g0[0] = 1u;                                           // count=1 (valid)
  g0[1] = lds;                                          // lds_addr (bytes)
  g0[2] = (u32)(ga & 0xffffffffu);                      // global_addr[31:0]
  g0[3] = (u32)((ga >> 32) & 0x1ffffffu) | (2u << 30);  // ga[56:32] | type=2
  int32x8 g1;
  // wg_mask=0 | data_size=2 (4B) | pad_enable | pad_interval=5 (64dw)
  // | pad_amount=3 (4dw)
  g1[0] = (int)((2u << 16) | (1u << 20) | (5u << 22) | (3u << 25));
  g1[1] = (int)(tile_w_dw << 16);                 // tensor_dim0[15:0]
  g1[2] = (int)((tile_w_dw >> 16) | (tile_h << 16));  // td0 hi | td1 lo
  g1[3] = (int)((tile_h >> 16) | (tile_w_dw << 16));  // td1 hi | tile_dim0
  g1[4] = (int)(tile_h & 0xffffu);                // tile_dim1 | tile_dim2=0
  g1[5] = (int)row_stride_dw;                     // tensor_dim0_stride[31:0]
  g1[6] = 0;                                      // stride hi | dim1_stride lo
  g1[7] = 0;
  int32x4 z4 = {0, 0, 0, 0};
#if defined(__clang_major__) && __clang_major__ >= 23
  int32x8 z8 = {0, 0, 0, 0, 0, 0, 0, 0};
  __builtin_amdgcn_tensor_load_to_lds(g0, g1, z4, z4, z8, 0);
#else
  __builtin_amdgcn_tensor_load_to_lds(g0, g1, z4, z4, 0);
#endif
}

// ------------------------------------------------------------ fp32 -> bf16
__global__ void cvt_kernel(const float* __restrict__ in, bf16* __restrict__ out,
                           int n) {
  int i = blockIdx.x * blockDim.x + threadIdx.x;
  int stride = gridDim.x * blockDim.x;
  for (; i < n; i += stride) out[i] = (bf16)in[i];
}

// fp32 [rows x cols] -> bf16 transposed [cols x rows]; rows = 1<<rshift
__global__ void cvt_t_kernel(const float* __restrict__ in,
                             bf16* __restrict__ out, int rshift, int cols) {
  int rows = 1 << rshift;
  int i = blockIdx.x * blockDim.x + threadIdx.x;
  int total = rows * cols;
  int stride = gridDim.x * blockDim.x;
  for (; i < total; i += stride) {
    int c = i >> rshift, r = i & (rows - 1);
    out[i] = (bf16)in[(size_t)r * cols + c];
  }
}

// ---------------------------------------------------------------- WMMA GEMM
// C[M x Nc] = A[M x Kd] * Bt^T, A row-major, Bt = B^T stored [Nc x Kd]
// (k-contiguous) so BOTH operand tiles stage with pure async b128 copies.
// 256 threads (8 waves), 128x128 tile, K-step 32, double-buffered LDS.
// Wave grid 4x2: wave w -> rows (w>>1)*32 (2 tiles), cols (w&1)*64 (4 tiles).
#define GEMM_TILE_ELEMS (128 * 40 + 128 * 40)
#define GEMM_SMEM (2 * GEMM_TILE_ELEMS * 2)

// stage a 128x32 k-contiguous tile (leading dim ld) into LDS [128][stride 40]
__device__ __forceinline__ void gemm_stage(const bf16* __restrict__ G, size_t ld,
                                           int row0, int k0, bf16* dst, int tid) {
  int r  = tid >> 1;
  int ks = (tid & 1) << 4;
  const bf16* g = G + (size_t)(row0 + r) * ld + k0 + ks;
  async_ld_b128(dst + r * 40 + ks, g);
  async_ld_b128(dst + r * 40 + ks + 8, g + 8);
}

__global__ __launch_bounds__(256) void gemm_bf16(
    const bf16* __restrict__ A, const bf16* __restrict__ Bt,
    float* __restrict__ Cf, bf16* __restrict__ Cb, bf16* __restrict__ Ct,
    int M, int Nc, int Kd) {
  extern __shared__ char smem[];
  bf16* buf0 = (bf16*)smem;                 // [As 128x40 | Bst 128x40] x2
  bf16* buf1 = buf0 + GEMM_TILE_ELEMS;

  const int tid  = threadIdx.x;
  const int wid  = tid >> 5;
  const int lane = tid & 31;
  const int hf   = (lane >> 4) & 1;
  const int l15  = lane & 15;
  const int bm = blockIdx.y * 128;
  const int bn = blockIdx.x * 128;
  const int wm = (wid >> 1) * 32;
  const int wn = (wid & 1) * 64;

  v8f acc[2][4];
#pragma unroll
  for (int a = 0; a < 2; ++a)
#pragma unroll
    for (int c = 0; c < 4; ++c)
#pragma unroll
      for (int r = 0; r < 8; ++r) acc[a][c][r] = 0.0f;

  // prologue: stage first tile (all async)
  gemm_stage(A,  Kd, bm, 0, buf0, tid);
  gemm_stage(Bt, Kd, bn, 0, buf0 + 128 * 40, tid);
  wait_async0();
  __syncthreads();

  int p = 0;
  for (int k0 = 0; k0 < Kd; k0 += 32) {
    bf16* cur = p ? buf1 : buf0;
    bf16* nxt = p ? buf0 : buf1;
    if (k0 + 32 < Kd) {  // async staging of next tile overlaps the WMMAs
      gemm_stage(A,  Kd, bm, k0 + 32, nxt, tid);
      gemm_stage(Bt, Kd, bn, k0 + 32, nxt + 128 * 40, tid);
    }
    if (k0 + 64 < Kd) {  // warm L2 one tile further ahead
      __builtin_prefetch(A + (size_t)(bm + (tid >> 1)) * Kd + k0 + 64, 0, 1);
      __builtin_prefetch(Bt + (size_t)(bn + (tid >> 1)) * Kd + k0 + 64, 0, 1);
    }
    bf16* As  = cur;
    bf16* Bst = cur + 128 * 40;
    v16bf af0 = frag_b128(As + (wm + l15) * 40, hf);
    v16bf af1 = frag_b128(As + (wm + 16 + l15) * 40, hf);
#pragma unroll
    for (int c = 0; c < 4; ++c) {
      v16bf bfv = frag_b128(Bst + (wn + c * 16 + l15) * 40, hf);
      acc[0][c] = wmma_bf16(af0, bfv, acc[0][c]);
      acc[1][c] = wmma_bf16(af1, bfv, acc[1][c]);
    }
    wait_async0();
    __syncthreads();
    p ^= 1;
  }

  // C/D layout: lane L holds col = L%16, VGPR r holds row = r + 8*(L/16)
#pragma unroll
  for (int a = 0; a < 2; ++a)
#pragma unroll
    for (int c = 0; c < 4; ++c)
#pragma unroll
      for (int r = 0; r < 8; ++r) {
        int row = bm + wm + a * 16 + r + 8 * hf;
        int col = bn + wn + c * 16 + l15;
        float vv = acc[a][c][r];
        if (Cf)      Cf[(size_t)row * Nc + col] = vv;
        else if (Cb) Cb[(size_t)row * Nc + col] = (bf16)vv;
        else         Ct[(size_t)col * M + row]  = (bf16)vv;  // column-major
      }
}

// ------------------------------------------------------- RMSNorm + RoPE (f32)
__global__ __launch_bounds__(256) void normrope_kernel(
    bf16* __restrict__ qb, bf16* __restrict__ kb, const float* __restrict__ qw,
    const float* __restrict__ kw, const int* __restrict__ segpos) {
  const int QHEADS = Bz * Ts * Nh;
  const int KHEADS = Bz * Ts * Kh;
  int idx = blockIdx.x * blockDim.x + threadIdx.x;
  if (idx >= QHEADS + KHEADS) return;
  bf16* base; const float* w; int t, b;
  if (idx < QHEADS) {
    base = qb + (size_t)idx * Hd; w = qw;
    b = idx / (Ts * Nh); t = (idx / Nh) % Ts;
  } else {
    int j = idx - QHEADS;
    base = kb + (size_t)j * Hd; w = kw;
    b = j / (Ts * Kh); t = (j / Kh) % Ts;
  }
  float x[Hd];
  const uint4* g = (const uint4*)base;
#pragma unroll
  for (int i = 0; i < Hd / 8; ++i) {
    union { uint4 u; bf16 h[8]; } tb; tb.u = g[i];
#pragma unroll
    for (int e = 0; e < 8; ++e) x[i * 8 + e] = (float)tb.h[e];
  }
  float ss = 0.f;
#pragma unroll
  for (int i = 0; i < Hd; ++i) ss += x[i] * x[i];
  float inv = rsqrtf(ss * (1.0f / Hd) + 1e-6f);
#pragma unroll
  for (int i = 0; i < Hd; ++i) x[i] = x[i] * inv * w[i];

  float pos = (float)segpos[b * Ts + t];
  float y[Hd];
#pragma unroll
  for (int i = 0; i < Hd / 2; ++i) {
    // timescale = 1e6^(i/64)  ->  angle = pos * exp(-ln(1e6)/64 * i)
    float ang = pos * __expf(-0.21586735f * (float)i);
    float sv, cv; __sincosf(ang, &sv, &cv);
    float fi = x[i], se = x[i + Hd / 2];
    y[i]          = fi * cv - se * sv;
    y[i + Hd / 2] = se * cv + fi * sv;
  }
  uint4* go = (uint4*)base;
#pragma unroll
  for (int i = 0; i < Hd / 8; ++i) {
    union { uint4 u; bf16 h[8]; } tb;
#pragma unroll
    for (int e = 0; e < 8; ++e) tb.h[e] = (bf16)y[i * 8 + e];
    go[i] = tb.u;
  }
}

// ------------------------------------------------------ flash attention (WMMA)
// grid (T/128, Nh, Bz); 256 threads = 8 waves; wave w owns q rows [w*16,+16).
// K is row-major (b,s,kv,h); V is pre-transposed (col-major from its GEMM):
// vt[(kv*Hd + h) * Mrows + b*Ts + s].  K/V tiles are fetched by the Tensor
// Data Mover (one DMA per tile, issued by wave 0, TENSORcnt-tracked) and
// double-buffered so the next s-tile streams in during S/softmax/PV.
#define ATTN_TILE (128 * 136)
#define ATTN_SMEM ((4 * ATTN_TILE + 8 * 16 * 136) * 2)

__global__ __launch_bounds__(256) void attn_kernel(
    const bf16* __restrict__ qb, const bf16* __restrict__ kb,
    const bf16* __restrict__ vt, bf16* __restrict__ ob) {
  extern __shared__ char smem[];
  bf16* KsB[2]; bf16* VtB[2];
  KsB[0] = (bf16*)smem;            // [s 128][h, stride 136]
  VtB[0] = KsB[0] + ATTN_TILE;     // [h 128][s, stride 136]
  KsB[1] = VtB[0] + ATTN_TILE;
  VtB[1] = KsB[1] + ATTN_TILE;
  bf16* Ps = VtB[1] + ATTN_TILE;   // per-wave [16][stride 136] P staging

  const int tid  = threadIdx.x;
  const int wid  = tid >> 5;
  const int lane = tid & 31;
  const int hf   = (lane >> 4) & 1;
  const int l15  = lane & 15;
  const int ttile = blockIdx.x;
  const int n  = blockIdx.y;
  const int b  = blockIdx.z;
  const int kv = n >> 1;  // n_rep = Nh/Kh = 2
  const int t0 = ttile * 128;
  const float scale = 0.08838834764831845f;  // 1/sqrt(128)

  const bf16* kbase = kb + ((size_t)(b * Ts) * Kh + kv) * Hd;  // + s*Kh*Hd
  const bf16* vbase = vt + (size_t)(kv * Hd) * Mrows + b * Ts; // + h*Mrows + s

  // Q fragments kept in registers for the whole kernel (A operand, 4 k-steps)
  v16bf qf[4];
  {
    int trow = t0 + wid * 16 + l15;
    const bf16* qbase = qb + ((size_t)(b * Ts + trow) * Nh + n) * Hd;
#pragma unroll
    for (int kk = 0; kk < 4; ++kk) qf[kk] = frag_b128(qbase + kk * 32, hf);
  }

  v8f Ov[8];
  float mrow[8], lrow[8];
#pragma unroll
  for (int c = 0; c < 8; ++c)
#pragma unroll
    for (int r = 0; r < 8; ++r) Ov[c][r] = 0.0f;
#pragma unroll
  for (int r = 0; r < 8; ++r) { mrow[r] = -1e30f; lrow[r] = 0.0f; }

  // prologue: TDM-stage first s-tile (wave 0 issues the two tile DMAs)
  if (wid == 0) {
    tdm_load_2d(KsB[0], kbase, Hd / 2, 128, (u32)(Kh * Hd / 2));
    tdm_load_2d(VtB[0], vbase, 128 / 2, 128, (u32)(Mrows / 2));
    __builtin_amdgcn_s_wait_tensorcnt((short)0);
  }
  __syncthreads();

  int p = 0;
  for (int st = 0; st <= ttile; ++st) {
    if (st + 1 <= ttile && wid == 0) {  // DMA next s-tile into other buffer
      int s1 = (st + 1) * 128;
      tdm_load_2d(KsB[1 - p], kbase + (size_t)s1 * Kh * Hd, Hd / 2, 128,
                  (u32)(Kh * Hd / 2));
      tdm_load_2d(VtB[1 - p], vbase + s1, 128 / 2, 128, (u32)(Mrows / 2));
    }
    bf16* Ks = KsB[p];
    bf16* Vt = VtB[p];

    // S = Q * K^T  (B operand lane = s column, k pairs contiguous in h)
    v8f Sacc[8];
#pragma unroll
    for (int c = 0; c < 8; ++c)
#pragma unroll
      for (int r = 0; r < 8; ++r) Sacc[c][r] = 0.0f;
#pragma unroll
    for (int c = 0; c < 8; ++c) {
      const bf16* bbase = Ks + (c * 16 + l15) * 136;
#pragma unroll
      for (int kk = 0; kk < 4; ++kk) {
        v16bf bfv = frag_b128(bbase + kk * 32, hf);
        Sacc[c] = wmma_bf16(qf[kk], bfv, Sacc[c]);
      }
    }

    // scale + analytic causal mask on the diagonal tile
#pragma unroll
    for (int c = 0; c < 8; ++c)
#pragma unroll
      for (int r = 0; r < 8; ++r) {
        float vv = Sacc[c][r] * scale;
        if (st == ttile) {
          int trl = wid * 16 + r + 8 * hf;
          int scl = c * 16 + l15;
          if (scl > trl) vv = -1e30f;
        }
        Sacc[c][r] = vv;
      }

    // online softmax: row lives in one 16-lane half -> butterfly over 1,2,4,8
    float mnew[8], alpha[8], rsum[8];
#pragma unroll
    for (int r = 0; r < 8; ++r) {
      float mx = Sacc[0][r];
#pragma unroll
      for (int c = 1; c < 8; ++c) mx = fmaxf(mx, Sacc[c][r]);
#pragma unroll
      for (int off = 1; off < 16; off <<= 1)
        mx = fmaxf(mx, __shfl_xor(mx, off, 32));
      mnew[r]  = fmaxf(mrow[r], mx);
      alpha[r] = __expf(mrow[r] - mnew[r]);
      rsum[r]  = 0.0f;
    }
#pragma unroll
    for (int c = 0; c < 8; ++c)
#pragma unroll
      for (int r = 0; r < 8; ++r) {
        float pe = __expf(Sacc[c][r] - mnew[r]);
        Sacc[c][r] = pe;
        rsum[r] += pe;
      }
#pragma unroll
    for (int r = 0; r < 8; ++r) {
#pragma unroll
      for (int off = 1; off < 16; off <<= 1)
        rsum[r] += __shfl_xor(rsum[r], off, 32);
      lrow[r] = lrow[r] * alpha[r] + rsum[r];
      mrow[r] = mnew[r];
    }
#pragma unroll
    for (int c = 0; c < 8; ++c)
#pragma unroll
      for (int r = 0; r < 8; ++r) Ov[c][r] *= alpha[r];

    // P -> per-wave LDS (bf16), then consumed as A fragments (wave-local)
    bf16* pw = Ps + wid * (16 * 136);
#pragma unroll
    for (int c = 0; c < 8; ++c)
#pragma unroll
      for (int r = 0; r < 8; ++r)
        pw[(r + 8 * hf) * 136 + c * 16 + l15] = (bf16)Sacc[c][r];

    // O += P * V  (V operand lane = h column, k pairs contiguous in s via Vt)
    const bf16* prow = pw + l15 * 136;
#pragma unroll
    for (int kk = 0; kk < 4; ++kk) {
      v16bf pf = frag_b128(prow + kk * 32, hf);
#pragma unroll
      for (int c = 0; c < 8; ++c) {
        v16bf vf = frag_b128(Vt + (c * 16 + l15) * 136 + kk * 32, hf);
        Ov[c] = wmma_bf16(pf, vf, Ov[c]);
      }
    }
    if (wid == 0) __builtin_amdgcn_s_wait_tensorcnt((short)0);
    __syncthreads();  // buffers swap: reads done + TDM writes landed
    p ^= 1;
  }

  // normalize and store (b,t,n,h) bf16
#pragma unroll
  for (int r = 0; r < 8; ++r) {
    float invl = 1.0f / lrow[r];
    int trow = t0 + wid * 16 + r + 8 * hf;
    bf16* od = ob + ((size_t)(b * Ts + trow) * Nh + n) * Hd;
#pragma unroll
    for (int c = 0; c < 8; ++c) od[c * 16 + l15] = (bf16)(Ov[c][r] * invl);
  }
}

// ---------------------------------------------------------------- launcher
extern "C" void kernel_launch(void* const* d_in, const int* in_sizes, int n_in,
                              void* d_out, int out_size, void* d_ws, size_t ws_size,
                              hipStream_t stream) {
  (void)in_sizes; (void)n_in; (void)out_size; (void)ws_size;
  const float* x  = (const float*)d_in[0];
  const int*   sp = (const int*)d_in[1];
  // d_in[2] = attn_mask: pure causal tril, reproduced analytically in-kernel
  const float* wq = (const float*)d_in[3];
  const float* wk = (const float*)d_in[4];
  const float* wv = (const float*)d_in[5];
  const float* wo = (const float*)d_in[6];
  const float* qw = (const float*)d_in[7];
  const float* kw = (const float*)d_in[8];
  float* out = (float*)d_out;

  // workspace layout (bf16 elements), total 71,303,168 bytes
  bf16* ws = (bf16*)d_ws;
  const size_t NX  = (size_t)Mrows * Dm;       // x
  const size_t NWQ = (size_t)Dm * Nh * Hd;     // wq^T / wo^T
  const size_t NWK = (size_t)Dm * Kh * Hd;     // wk^T / wv^T
  const size_t NQ  = (size_t)Mrows * Nh * Hd;  // q / attn-out
  const size_t NK  = (size_t)Mrows * Kh * Hd;  // k / v^T
  bf16* xb  = ws;
  bf16* wqt = xb  + NX;   // [Nh*Hd x Dm]  (k-contiguous)
  bf16* wkt = wqt + NWQ;  // [Kh*Hd x Dm]
  bf16* wvt = wkt + NWK;  // [Kh*Hd x Dm]
  bf16* wot = wvt + NWK;  // [Dm x Nh*Hd]
  bf16* qbp = wot + NWQ;  // (b,t,n,h) row-major
  bf16* kbp = qbp + NQ;   // (b,s,kv,h) row-major
  bf16* vtp = kbp + NK;   // V^T column-major: [(kv*Hd+h) x Mrows]
  bf16* obp = vtp + NK;   // attention out (b,t,n,h) row-major

  cvt_kernel  <<<1024, 256, 0, stream>>>(x, xb, (int)NX);
  cvt_t_kernel<<<1024, 256, 0, stream>>>(wq, wqt, 10, Nh * Hd);  // rows=1024
  cvt_t_kernel<<<1024, 256, 0, stream>>>(wk, wkt, 10, Kh * Hd);
  cvt_t_kernel<<<1024, 256, 0, stream>>>(wv, wvt, 10, Kh * Hd);
  cvt_t_kernel<<<1024, 256, 0, stream>>>(wo, wot, 11, Dm);       // rows=2048

  // projections: q (4096x2048), k (4096x1024), v^T (1024x4096 col-major out)
  gemm_bf16<<<dim3(16, 32), 256, GEMM_SMEM, stream>>>(
      xb, wqt, nullptr, qbp, nullptr, Mrows, Nh * Hd, Dm);
  gemm_bf16<<<dim3(8, 32), 256, GEMM_SMEM, stream>>>(
      xb, wkt, nullptr, kbp, nullptr, Mrows, Kh * Hd, Dm);
  gemm_bf16<<<dim3(8, 32), 256, GEMM_SMEM, stream>>>(
      xb, wvt, nullptr, nullptr, vtp, Mrows, Kh * Hd, Dm);

  normrope_kernel<<<(Bz * Ts * (Nh + Kh) + 255) / 256, 256, 0, stream>>>(
      qbp, kbp, qw, kw, sp);

  attn_kernel<<<dim3(Ts / 128, Nh, Bz), 256, ATTN_SMEM, stream>>>(qbp, kbp, vtp,
                                                                  obp);

  // output projection: (4096x2048) x wo -> fp32 d_out (4096x1024)
  gemm_bf16<<<dim3(Dm / 128, 32), 256, GEMM_SMEM, stream>>>(
      obp, wot, out, nullptr, nullptr, Mrows, Dm, Nh * Hd);
}
